// QLinearExperimentalCPU_59425167507995
// MI455X (gfx1250) — compile-verified, matrix operand order checked by reading, not run on hardware
//
#include <hip/hip_runtime.h>

typedef int v8i __attribute__((ext_vector_type(8)));

// Problem shape (fixed by the reference)
constexpr int Bdim = 4;
constexpr int S    = 4096;
constexpr int K    = 1024;
constexpr int N    = 4096;

// GEMM tiling
constexpr int MT = 128;   // rows (S) per workgroup
constexpr int NT = 128;   // cols (N) per workgroup
constexpr int KT = 64;    // K per step == WMMA K

// ---------------------------------------------------------------------------
// CDNA5 async global->LDS copy (16B per lane), ASYNCcnt-tracked.
// lds  = wave-relative LDS byte address (low 32 bits of generic shared ptr)
// ga   = 64-bit global base address (loop-invariant)
// ioff = compile-time signed 24-bit instruction offset (k0 folded in here)
// ---------------------------------------------------------------------------
__device__ __forceinline__ void async_ld16(unsigned lds, unsigned long long ga,
                                           int ioff) {
    asm volatile("global_load_async_to_lds_b128 %0, %1, off offset:%2"
                 :: "v"(lds), "v"(ga), "i"(ioff) : "memory");
}
__device__ __forceinline__ void wait_async0() {
    asm volatile("s_wait_asynccnt 0x0" ::: "memory");
}

// ---------------------------------------------------------------------------
// Pass 0: zero the per-batch absmax bit buckets
// ---------------------------------------------------------------------------
__global__ void init_scales_kernel(unsigned* __restrict__ smax) {
    if (threadIdx.x < Bdim) smax[threadIdx.x] = 0u;
}

// ---------------------------------------------------------------------------
// Pass 1: per-batch-slice absmax. |x| >= 0 so the IEEE bit pattern is
// monotonic -> integer atomicMax is exact.
// ---------------------------------------------------------------------------
__global__ void absmax_kernel(const float* __restrict__ x,
                              unsigned* __restrict__ smax) {
    const int b = blockIdx.y;
    const float4* xb = (const float4*)(x + (size_t)b * S * K);
    const int n4 = (S * K) / 4;
    float m = 0.0f;
    for (int i = blockIdx.x * blockDim.x + threadIdx.x; i < n4;
         i += gridDim.x * blockDim.x) {
        float4 v = xb[i];
        m = fmaxf(m, fmaxf(fmaxf(fabsf(v.x), fabsf(v.y)),
                           fmaxf(fabsf(v.z), fabsf(v.w))));
    }
    for (int off = 16; off > 0; off >>= 1)
        m = fmaxf(m, __shfl_xor(m, off, 32));
    if ((threadIdx.x & 31) == 0)
        atomicMax(&smax[b], __float_as_uint(m));
}

// ---------------------------------------------------------------------------
// Pass 2: quantize x -> int8 (packed 4/word). round-half-even via rintf to
// match jnp.round.
// ---------------------------------------------------------------------------
__global__ void quant_kernel(const float* __restrict__ x,
                             const unsigned* __restrict__ smax,
                             int* __restrict__ xq4) {
    const int b = blockIdx.y;
    const float scale = __uint_as_float(smax[b]) * (1.0f / 128.0f);
    const float inv = 1.0f / scale;
    const float4* xb = (const float4*)(x + (size_t)b * S * K);
    int* qb = xq4 + (size_t)b * ((S * K) / 4);
    const int i = blockIdx.x * blockDim.x + threadIdx.x;
    float4 v = xb[i];
    int q0 = (int)fminf(fmaxf(rintf(v.x * inv), -128.0f), 127.0f);
    int q1 = (int)fminf(fmaxf(rintf(v.y * inv), -128.0f), 127.0f);
    int q2 = (int)fminf(fmaxf(rintf(v.z * inv), -128.0f), 127.0f);
    int q3 = (int)fminf(fmaxf(rintf(v.w * inv), -128.0f), 127.0f);
    qb[i] = (q0 & 255) | ((q1 & 255) << 8) | ((q2 & 255) << 16) | ((q3 & 255) << 24);
}

// ---------------------------------------------------------------------------
// Pass 3: weight pack + transpose: int32 [K,N] -> int8 wT[N,K] so WMMA
// B-fragments read contiguous K bytes. LDS-tiled 32x32 transpose.
// ---------------------------------------------------------------------------
__global__ __launch_bounds__(256) void wpack_kernel(const int* __restrict__ wq,
                                                    signed char* __restrict__ wT) {
    __shared__ signed char tile[32][36];  // [k_local][n_local], padded
    const int n0 = blockIdx.x * 32;
    const int k0 = blockIdx.y * 32;
    const int tx = threadIdx.x;           // 0..31 -> n
    const int ty = threadIdx.y;           // 0..7
    for (int r = ty; r < 32; r += 8)
        tile[r][tx] = (signed char)wq[(size_t)(k0 + r) * N + n0 + tx];
    __syncthreads();
    const int t = ty * 32 + tx;           // 0..255
    const int n = t >> 3;                 // 0..31
    const int c = t & 7;                  // 0..7 -> 4 k-bytes each
    unsigned p = ((unsigned)(unsigned char)tile[4 * c + 0][n]) |
                 ((unsigned)(unsigned char)tile[4 * c + 1][n] << 8) |
                 ((unsigned)(unsigned char)tile[4 * c + 2][n] << 16) |
                 ((unsigned)(unsigned char)tile[4 * c + 3][n] << 24);
    *(unsigned*)(wT + (size_t)(n0 + n) * K + k0 + 4 * c) = p;
}

// ---------------------------------------------------------------------------
// Pass 4: iu8 WMMA GEMM, double-buffered async global->LDS staging with
// immediate-offset addressing. All 6 fragments (4xA, 2xB) are pre-loaded into
// distinct registers before the 8-WMMA block to keep ds writes out of the
// WMMA_IU8 8-slot WAR hazard window. Fused requant/dequant/bias epilogue
// with non-temporal stores.
// Workgroup: 256 threads = 8 waves; tile 128x128, K-step 64.
// Wave (wm,wn) = (wave&1, wave>>1) owns 64 rows x 32 cols = 4x2 WMMA tiles.
// ---------------------------------------------------------------------------
__global__ __launch_bounds__(256) void gemm_iu8_kernel(
    const signed char* __restrict__ xq, const signed char* __restrict__ wT,
    const unsigned* __restrict__ smax, const float* __restrict__ bias,
    float* __restrict__ out) {
    const int b  = blockIdx.z;
    const int m0 = blockIdx.y * MT;
    const int n0 = blockIdx.x * NT;
    const signed char* A = xq + (size_t)b * S * K;

    const float x_scale = __uint_as_float(smax[b]) * (1.0f / 128.0f);
    const float dq = x_scale * 0.25f;  // x_scale * Y_SCALE * REQ_OUT

    // Double-buffered tiles: addr = row*64 + k within a buffer.
    __shared__ __align__(16) signed char lA[2][MT * KT];  // 2 x 8 KB
    __shared__ __align__(16) signed char lB[2][NT * KT];  // 2 x 8 KB

    const int t    = threadIdx.x;
    const int lane = t & 31;
    const int wave = t >> 5;
    const int wm   = wave & 1;
    const int wn   = wave >> 1;
    const int half = lane >> 4;   // 8-bit-A K-interleave half
    const int lr   = lane & 15;   // M (A) / N (B,C,D) within tile

    // Loop-invariant per-thread global bases: chunk c (of 512) = 16B at
    // row c>>2, seg c&3; this thread moves chunks t and t+256 of each tile.
    // k0 goes into the instruction's 24-bit immediate offset.
    const unsigned long long gA0 =
        (unsigned long long)(A + (size_t)(m0 + (t >> 2)) * K + (t & 3) * 16);
    const unsigned long long gA1 = gA0 + (unsigned long long)64 * K;
    const unsigned long long gB0 =
        (unsigned long long)(wT + (size_t)(n0 + (t >> 2)) * K + (t & 3) * 16);
    const unsigned long long gB1 = gB0 + (unsigned long long)64 * K;

    const unsigned baseA = (unsigned)(uintptr_t)&lA[0][0];
    const unsigned baseB = (unsigned)(uintptr_t)&lB[0][0];
    const unsigned cOff0 = (unsigned)t * 16u;
    const unsigned cOff1 = cOff0 + 4096u;

    v8i acc[4][2] = {};

    // Prologue: async-load tile k0=0 into buffer 0.
    async_ld16(baseA + cOff0, gA0, 0);
    async_ld16(baseA + cOff1, gA1, 0);
    async_ld16(baseB + cOff0, gB0, 0);
    async_ld16(baseB + cOff1, gB1, 0);
    wait_async0();
    __syncthreads();

#pragma unroll
    for (int kt = 0; kt < K / KT; ++kt) {
        const int cur = kt & 1;
        const unsigned bufSel = (unsigned)(cur ^ 1) * 8192u;
        if (kt + 1 < K / KT) {  // prefetch next tile while computing this one
            const int koff = (kt + 1) * KT;  // compile-time after unroll
            async_ld16(baseA + bufSel + cOff0, gA0, koff);
            async_ld16(baseA + bufSel + cOff1, gA1, koff);
            async_ld16(baseB + bufSel + cOff0, gB0, koff);
            async_ld16(baseB + bufSel + cOff1, gB1, koff);
        }

        const signed char* bufA = &lA[cur][0];
        const signed char* bufB = &lB[cur][0];

        // Pre-load ALL fragments into distinct registers, then issue the
        // 8 WMMAs: keeps ds_load writes outside the IU8 WAR hazard window.
        v8i bf[2];
#pragma unroll
        for (int j = 0; j < 2; ++j) {
            const signed char* bp = bufB + (wn * 32 + j * 16 + lr) * KT + half * 8;
            uint2 p0 = *(const uint2*)(bp + 0);
            uint2 p1 = *(const uint2*)(bp + 16);
            uint2 p2 = *(const uint2*)(bp + 32);
            uint2 p3 = *(const uint2*)(bp + 48);
            bf[j] = (v8i){(int)p0.x, (int)p0.y, (int)p1.x, (int)p1.y,
                          (int)p2.x, (int)p2.y, (int)p3.x, (int)p3.y};
        }
        v8i af[4];
#pragma unroll
        for (int i = 0; i < 4; ++i) {
            const signed char* ap = bufA + (wm * 64 + i * 16 + lr) * KT + half * 8;
            uint2 q0 = *(const uint2*)(ap + 0);
            uint2 q1 = *(const uint2*)(ap + 16);
            uint2 q2 = *(const uint2*)(ap + 32);
            uint2 q3 = *(const uint2*)(ap + 48);
            af[i] = (v8i){(int)q0.x, (int)q0.y, (int)q1.x, (int)q1.y,
                          (int)q2.x, (int)q2.y, (int)q3.x, (int)q3.y};
        }
#pragma unroll
        for (int i = 0; i < 4; ++i) {
#pragma unroll
            for (int j = 0; j < 2; ++j) {
                acc[i][j] = __builtin_amdgcn_wmma_i32_16x16x64_iu8(
                    true, af[i], true, bf[j], acc[i][j], false, false);
            }
        }

        // Ensure this wave's prefetch landed, then barrier so every wave's
        // prefetch (and every wave's reads of `cur`) are done before swap.
        wait_async0();
        __syncthreads();
    }

    // Epilogue: round(acc/16) clipped to int16 -> *dq + bias; NT stores.
    // C/D layout: VGPR r holds M = r + 8*half, N = lr.
    float* outb = out + (size_t)b * S * N;
#pragma unroll
    for (int i = 0; i < 4; ++i) {
#pragma unroll
        for (int j = 0; j < 2; ++j) {
            const int tm = m0 + wm * 64 + i * 16;
            const int tn = n0 + wn * 32 + j * 16 + lr;
            const float bv = bias[tn];
#pragma unroll
            for (int r = 0; r < 8; ++r) {
                const int M = tm + r + half * 8;
                float y16 = fminf(
                    fmaxf(rintf((float)acc[i][j][r] * 0.0625f), -32768.0f),
                    32767.0f);
                __builtin_nontemporal_store(y16 * dq + bv,
                                            &outb[(size_t)M * N + tn]);
            }
        }
    }
}

// ---------------------------------------------------------------------------
extern "C" void kernel_launch(void* const* d_in, const int* in_sizes, int n_in,
                              void* d_out, int out_size, void* d_ws, size_t ws_size,
                              hipStream_t stream) {
    const float* x    = (const float*)d_in[0];   // [B,S,K] fp32
    const int*   wq   = (const int*)d_in[1];     // [K,N] int32 (int8 range)
    const float* bias = (const float*)d_in[2];   // [N] fp32
    float* out = (float*)d_out;                  // [B,S,N] fp32

    // Workspace: [0,256) scale bits; then 16MB xq int8; then 4MB wT int8.
    unsigned* smax = (unsigned*)d_ws;
    signed char* xq = (signed char*)d_ws + 256;
    signed char* wT = xq + (size_t)Bdim * S * K;

    init_scales_kernel<<<1, 32, 0, stream>>>(smax);
    absmax_kernel<<<dim3(256, Bdim), 256, 0, stream>>>(x, smax);
    quant_kernel<<<dim3((S * K / 4) / 256, Bdim), 256, 0, stream>>>(
        x, smax, (int*)xq);
    wpack_kernel<<<dim3(N / 32, K / 32), dim3(32, 8), 0, stream>>>(wq, wT);
    gemm_iu8_kernel<<<dim3(N / NT, S / MT, Bdim), 256, 0, stream>>>(
        xq, wT, smax, bias, out);
}
